// GraphConvClf_18683107737858
// MI455X (gfx1250) — compile-verified, matrix operand order checked by reading, not run on hardware
//
#include <hip/hip_runtime.h>

typedef __attribute__((ext_vector_type(16))) _Float16 v16h;
typedef __attribute__((ext_vector_type(8)))  _Float16 v8h;
typedef __attribute__((ext_vector_type(8)))  float    v8f;

#define NV_   160000
#define NE_   480000
#define BATCH 16
#define VPG   10000
#define FC1D  1024
#define NCLS  10

// halfs between W0 and W1 in the packed weight buffer (fixed, max-size slot)
#define WOFF  (256 * 256)

// ---------------- convert f32 -> f16 with optional relu and K zero-padding ----
__global__ void cvt_relu_pad(const float* __restrict__ src, _Float16* __restrict__ dst,
                             int total, int K, int kpshift, int do_relu) {
  int idx = blockIdx.x * blockDim.x + threadIdx.x;
  if (idx >= total) return;
  int k = idx & ((1 << kpshift) - 1);
  int r = idx >> kpshift;
  float v = 0.f;
  if (k < K) {
    v = src[(size_t)r * K + k];
    if (do_relu) v = v > 0.f ? v : 0.f;
  }
  dst[idx] = (_Float16)v;
}

// ---------------- dual GEMM: self = X@W0^T + b0, neigh = X@W1^T + b1 ----------
// X: (N, K) f16 row-major.  W: packed weights, W0 at +0, W1 at +WOFF halfs,
// both (M, K) row-major.  K, M compile-time -> all tile offsets fold into
// global_load immediates; only two live base pointers in the K-loop.
// One wave: 16 rows x 2 col tiles x 2 matrices = 4 WMMA per A fragment.
#define COLT 2
template <int K, int M>
__global__ void gc_dual_gemm(const _Float16* __restrict__ X,
                             const _Float16* __restrict__ W,
                             const float* __restrict__ b0,
                             const float* __restrict__ b1,
                             float* __restrict__ self_out,
                             _Float16* __restrict__ neigh_out) {
  const int lane = threadIdx.x & 31;
  const int wave = threadIdx.x >> 5;
  const int row0 = (blockIdx.x * 8 + wave) * 16;     // vertex rows
  const int col0 = blockIdx.y * (16 * COLT);         // output features
  const int mrow = lane & 15;
  const int half = lane >> 4;

  // A-fragment: lane holds row (row0+mrow); v16h element i covers
  // K = 8*half + i (i<8) and 16 + 8*half + (i-8) -> two 16B loads per K-step.
  const _Float16* xrow = X + (size_t)(row0 + mrow) * K + 8 * half;
  const _Float16* wrow = W + (size_t)(col0 + mrow) * K + 8 * half;

  v8f c0[COLT], c1[COLT];
#pragma unroll
  for (int t = 0; t < COLT; ++t) { c0[t] = (v8f){}; c1[t] = (v8f){}; }

  for (int k0 = 0; k0 < K; k0 += 32) {
    if (k0 + 32 < K) {
      __builtin_prefetch(xrow + k0 + 32, 0, 3);      // global_prefetch_b8
    }
    v8h alo = *(const v8h*)(xrow + k0);
    v8h ahi = *(const v8h*)(xrow + k0 + 16);
    v16h a;
#pragma unroll
    for (int i = 0; i < 8; ++i) { a[i] = alo[i]; a[i + 8] = ahi[i]; }

#pragma unroll
    for (int t = 0; t < COLT; ++t) {
      const _Float16* wt = wrow + t * (16 * K) + k0;  // offsets fold to immediates
      v8h l0 = *(const v8h*)(wt);
      v8h h0 = *(const v8h*)(wt + 16);
      v8h l1 = *(const v8h*)(wt + WOFF);
      v8h h1 = *(const v8h*)(wt + WOFF + 16);
      v16h f0, f1;
#pragma unroll
      for (int i = 0; i < 8; ++i) {
        f0[i] = l0[i]; f0[i + 8] = h0[i];
        f1[i] = l1[i]; f1[i + 8] = h1[i];
      }
      c0[t] = __builtin_amdgcn_wmma_f32_16x16x32_f16(false, a, false, f0, (short)0, c0[t], false, false);
      c1[t] = __builtin_amdgcn_wmma_f32_16x16x32_f16(false, a, false, f1, (short)0, c1[t], false, false);
    }
  }

  // D layout: lane n = lane&15 is output feature; VGPR v -> row v + 8*half
  const int n = lane & 15;
#pragma unroll
  for (int t = 0; t < COLT; ++t) {
    const int c = col0 + t * 16 + n;
    const float bias0 = b0[c];
    const float bias1 = b1[c];
#pragma unroll
    for (int v = 0; v < 8; ++v) {
      int r = row0 + v + 8 * half;
      self_out [(size_t)r * M + c] = c0[t][v] + bias0;
      neigh_out[(size_t)r * M + c] = (_Float16)(c1[t][v] + bias1);
    }
  }
}

// ---------------- edge scatter: out[i0] += neigh[i1]; out[i1] += neigh[i0] ----
// neigh is f16 (halves the dominant random-read traffic); out stays f32 and is
// updated with hardware global_atomic_add_f32. Feature index fastest-varying.
__global__ void edge_scatter(const int* __restrict__ edges,
                             const _Float16* __restrict__ neigh,
                             float* __restrict__ out, int mshift) {
  unsigned tid = blockIdx.x * blockDim.x + threadIdx.x;
  unsigned total = (unsigned)NE_ << mshift;
  if (tid >= total) return;
  int f = (int)(tid & ((1u << mshift) - 1));
  int e = (int)(tid >> mshift);
  int i0 = edges[2 * e];
  int i1 = edges[2 * e + 1];
  float v01 = (float)neigh[((size_t)i1 << mshift) + f];
  float v10 = (float)neigh[((size_t)i0 << mshift) + f];
  unsafeAtomicAdd(&out[((size_t)i0 << mshift) + f], v01);
  unsafeAtomicAdd(&out[((size_t)i1 << mshift) + f], v10);
}

// ---------------- pooling (with fused final relu): mean over 10000 verts ------
__global__ void zero_f32(float* __restrict__ p, int n) {
  int i = blockIdx.x * blockDim.x + threadIdx.x;
  if (i < n) p[i] = 0.f;
}

#define POOL_CHUNKS 20
#define POOL_ROWS   (VPG / POOL_CHUNKS)   // 500
__global__ void pool_relu_sum(const float* __restrict__ x, float* __restrict__ pooled) {
  int b = blockIdx.x / POOL_CHUNKS;
  int chunk = blockIdx.x % POOL_CHUNKS;
  int c = threadIdx.x;                    // 0..255 feature
  const float* base = x + ((size_t)b * VPG + (size_t)chunk * POOL_ROWS) * 256 + c;
  float s = 0.f;
  for (int r = 0; r < POOL_ROWS; ++r)
    s += fmaxf(base[(size_t)r * 256], 0.f);
  unsafeAtomicAdd(&pooled[b * 256 + c], s);
}

// ---------------- FC head (tiny, scalar f32) ----------------------------------
__global__ void fc1_kernel(const float* __restrict__ pooled,
                           const float* __restrict__ w, const float* __restrict__ b,
                           float* __restrict__ h) {
  int o = blockIdx.x * blockDim.x + threadIdx.x;
  if (o >= BATCH * FC1D) return;
  int bi = o / FC1D;
  int j  = o % FC1D;
  const float inv = 1.0f / (float)VPG;
  float s = b[j];
  const float* pr = pooled + bi * 256;
  const float* wr = w + (size_t)j * 256;
  for (int k = 0; k < 256; ++k) s += pr[k] * inv * wr[k];
  h[o] = s > 0.f ? s : 0.f;
}

__global__ void fc2_kernel(const float* __restrict__ h,
                           const float* __restrict__ w, const float* __restrict__ b,
                           float* __restrict__ out) {
  int o = blockIdx.x * blockDim.x + threadIdx.x;
  if (o >= BATCH * NCLS) return;
  int bi = o / NCLS;
  int j  = o % NCLS;
  float s = b[j];
  const float* hr = h + bi * FC1D;
  const float* wr = w + (size_t)j * FC1D;
  for (int k = 0; k < FC1D; ++k) s += hr[k] * wr[k];
  out[o] = s;
}

// ------------------------------------------------------------------------------
extern "C" void kernel_launch(void* const* d_in, const int* in_sizes, int n_in,
                              void* d_out, int out_size, void* d_ws, size_t ws_size,
                              hipStream_t stream) {
  const float* verts = (const float*)d_in[0];
  const int*   edges = (const int*)d_in[1];
  const float* gw0[3] = { (const float*)d_in[2],  (const float*)d_in[6],  (const float*)d_in[10] };
  const float* gb0[3] = { (const float*)d_in[3],  (const float*)d_in[7],  (const float*)d_in[11] };
  const float* gw1[3] = { (const float*)d_in[4],  (const float*)d_in[8],  (const float*)d_in[12] };
  const float* gb1[3] = { (const float*)d_in[5],  (const float*)d_in[9],  (const float*)d_in[13] };
  const float* fc1w = (const float*)d_in[14];
  const float* fc1b = (const float*)d_in[15];
  const float* fc2w = (const float*)d_in[16];
  const float* fc2b = (const float*)d_in[17];
  float* outp = (float*)d_out;

  // workspace carve-up (all offsets 256B aligned)
  char* ws = (char*)d_ws;
  size_t off = 0;
  _Float16* xb    = (_Float16*)(ws + off); off += (size_t)NV_ * 256 * sizeof(_Float16); // 81.92 MB
  float*    selfb = (float*)(ws + off);    off += (size_t)NV_ * 256 * sizeof(float);    // 163.84 MB
  _Float16* neigh = (_Float16*)(ws + off); off += (size_t)NV_ * 256 * sizeof(_Float16); // 81.92 MB
  _Float16* wb0   = (_Float16*)(ws + off); off += (size_t)WOFF * sizeof(_Float16);      // W0 slot
  _Float16* wb1   = (_Float16*)(ws + off); off += (size_t)WOFF * sizeof(_Float16);      // W1 = W0 + WOFF
  float*    pooled= (float*)(ws + off);    off += (size_t)BATCH * 256 * sizeof(float);
  float*    hbuf  = (float*)(ws + off);    off += (size_t)BATCH * FC1D * sizeof(float);

  const int din   [3] = { 3,  128, 256 };   // actual input dims
  const int dinp  [3] = { 32, 128, 256 };   // padded (mult of 32) for WMMA K
  const int kshift[3] = { 5,  7,   8   };   // log2(dinp)
  const int dout  [3] = { 128, 256, 256 };
  const int mshift[3] = { 7,  8,   8   };   // log2(dout)

  for (int l = 0; l < 3; ++l) {
    // 1) convert/pad activations (relu fused for l>0) and weights to f16
    {
      const float* src = (l == 0) ? verts : selfb;
      int total = NV_ * dinp[l];
      cvt_relu_pad<<<(total + 255) / 256, 256, 0, stream>>>(src, xb, total, din[l], kshift[l], l > 0);
      int wtotal = dout[l] * dinp[l];
      cvt_relu_pad<<<(wtotal + 255) / 256, 256, 0, stream>>>(gw0[l], wb0, wtotal, din[l], kshift[l], 0);
      cvt_relu_pad<<<(wtotal + 255) / 256, 256, 0, stream>>>(gw1[l], wb1, wtotal, din[l], kshift[l], 0);
    }
    // 2) dual GEMM: self + neigh in one pass over x, 2 col tiles per wave
    {
      dim3 grid(NV_ / 128, dout[l] / (16 * COLT));
      if (l == 0)
        gc_dual_gemm<32, 128><<<grid, 256, 0, stream>>>(xb, wb0, gb0[l], gb1[l], selfb, neigh);
      else if (l == 1)
        gc_dual_gemm<128, 256><<<grid, 256, 0, stream>>>(xb, wb0, gb0[l], gb1[l], selfb, neigh);
      else
        gc_dual_gemm<256, 256><<<grid, 256, 0, stream>>>(xb, wb0, gb0[l], gb1[l], selfb, neigh);
    }
    // 3) edge scatter-add into selfb (hardware global_atomic_add_f32)
    {
      unsigned total = (unsigned)NE_ << mshift[l];
      edge_scatter<<<(total + 255) / 256, 256, 0, stream>>>(edges, neigh, selfb, mshift[l]);
    }
  }

  // 4) mean pool (fused final relu)
  zero_f32<<<(BATCH * 256 + 255) / 256, 256, 0, stream>>>(pooled, BATCH * 256);
  pool_relu_sum<<<BATCH * POOL_CHUNKS, 256, 0, stream>>>(selfb, pooled);

  // 5) FC head
  fc1_kernel<<<(BATCH * FC1D + 255) / 256, 256, 0, stream>>>(pooled, fc1w, fc1b, hbuf);
  fc2_kernel<<<1, 256, 0, stream>>>(hbuf, fc2w, fc2b, outp);
}
